// TransformerBlock_34909494182367
// MI455X (gfx1250) — matrix operand order repaired and checked by reading, not compiled, not run on hardware
//
#include <hip/hip_runtime.h>
#include <math.h>

// ---------------------------------------------------------------------------
// Transformer block for MI455X (gfx1250), bf16 WMMA everywhere.
//   B=4, N=2048, DIM=1024, HEADS=16, HEAD_DIM=64, HIDDEN=4096, M = B*N = 8192
// ---------------------------------------------------------------------------

#define Bb     4
#define Nn     2048
#define DIMd   1024
#define HEADS  16
#define HD     64
#define HID    4096
#define Mrows  (Bb*Nn)          // 8192
#define NQKV   (3*DIMd)         // 3072

typedef unsigned short bf16_t;  // storage type
typedef __attribute__((ext_vector_type(16))) __bf16 v16bf;
typedef __attribute__((ext_vector_type(8)))  __bf16 v8bf;
typedef __attribute__((ext_vector_type(8)))  float  v8f;
typedef __attribute__((ext_vector_type(4)))  int    v4i;

__device__ __forceinline__ bf16_t f2bf(float f) {
    union { float f; unsigned u; } x; x.f = f;
    unsigned u = x.u;
    unsigned r = (u + 0x7FFFu + ((u >> 16) & 1u)) >> 16;   // RNE
    return (bf16_t)r;
}
__device__ __forceinline__ v16bf join16(v8bf lo, v8bf hi) {
    union { v16bf v; v8bf h[2]; } u;
    u.h[0] = lo; u.h[1] = hi;
    return u.v;
}
// A operand (16x32 bf16): lane L (m = L&15): halves 0..15 -> k = 16*(L>>4)+i
//
__device__ __forceinline__ v16bf load_a16x32(const bf16_t* row, int kk, int hi16) {
    v8bf lo = *reinterpret_cast<const v8bf*>(row + kk + 16 * hi16);
    v8bf hh = *reinterpret_cast<const v8bf*>(row + kk + 16 * hi16 + 8);
    return join16(lo, hh);
}

#define WMMA_BF16(a, bm, c) \
    __builtin_amdgcn_wmma_f32_16x16x32_bf16(false, (a), false, (bm), (short)0, (c), false, false)

// ---- CDNA5 async global->LDS copy (ASYNCcnt-tracked DMA), with fallback ----
#if __has_builtin(__builtin_amdgcn_global_load_async_to_lds_b128)
#define HAVE_ASYNC_LDS 1
#else
#define HAVE_ASYNC_LDS 0
#endif

__device__ __forceinline__ void copy16_g2l(const bf16_t* g, bf16_t* l) {
#if HAVE_ASYNC_LDS
    __builtin_amdgcn_global_load_async_to_lds_b128(
        (__attribute__((address_space(1))) v4i*)g,
        (__attribute__((address_space(3))) v4i*)l, 0, 0);
#else
    *reinterpret_cast<v8bf*>(l) = *reinterpret_cast<const v8bf*>(g);
#endif
}
__device__ __forceinline__ void wait_async_le1() {
#if HAVE_ASYNC_LDS
#if __has_builtin(__builtin_amdgcn_s_wait_asynccnt)
    __builtin_amdgcn_s_wait_asynccnt(1);
#else
    asm volatile("s_wait_asynccnt 0x1" ::: "memory");
#endif
#endif
}
__device__ __forceinline__ void wait_async_le0() {
#if HAVE_ASYNC_LDS
#if __has_builtin(__builtin_amdgcn_s_wait_asynccnt)
    __builtin_amdgcn_s_wait_asynccnt(0);
#else
    asm volatile("s_wait_asynccnt 0x0" ::: "memory");
#endif
#endif
}

// ---------------------------------------------------------------------------
// weight convert + transpose:  wT[n*K + k] = bf16(w[k*Ncol + n])
// ---------------------------------------------------------------------------
__global__ void cvtT_kernel(const float* __restrict__ w, bf16_t* __restrict__ wT,
                            int K, int Ncol) {
    size_t idx = (size_t)blockIdx.x * 256 + threadIdx.x;
    size_t tot = (size_t)K * Ncol;
    if (idx >= tot) return;
    int k = (int)(idx / Ncol);
    int n = (int)(idx % Ncol);
    wT[(size_t)n * K + k] = f2bf(w[idx]);
}

// ---------------------------------------------------------------------------
// row-wise layernorm -> bf16 (one block of 256 per row, D = 1024)
// ---------------------------------------------------------------------------
__global__ void ln_bf16_kernel(const float* __restrict__ x, const float* __restrict__ g,
                               const float* __restrict__ bta, bf16_t* __restrict__ out, int D) {
    __shared__ float s1[256], s2[256];
    const int row = blockIdx.x;
    const float* xr = x + (size_t)row * D;
    float sum = 0.f, sq = 0.f;
    for (int i = threadIdx.x; i < D; i += 256) { float v = xr[i]; sum += v; sq += v * v; }
    s1[threadIdx.x] = sum; s2[threadIdx.x] = sq;
    __syncthreads();
    for (int s = 128; s > 0; s >>= 1) {
        if (threadIdx.x < s) { s1[threadIdx.x] += s1[threadIdx.x + s]; s2[threadIdx.x] += s2[threadIdx.x + s]; }
        __syncthreads();
    }
    float mu = s1[0] / D;
    float var = s2[0] / D - mu * mu;
    float rs = rsqrtf(var + 1e-5f);
    for (int i = threadIdx.x; i < D; i += 256)
        out[(size_t)row * D + i] = f2bf((xr[i] - mu) * rs * g[i] + bta[i]);
}

// ---------------------------------------------------------------------------
// Workgroup-tiled bf16 GEMM:  C[M,Nout] = A[M,K] @ B[K,Nout]  (BT[Nout,K] given)
// Block = 256 threads (8 waves). Block tile: 128(M) x 64(N); wave w owns rows
// [m0+16w, m0+16w+16). The 64x32 B tile for each K-step is staged into LDS by
// async global->LDS DMA (ASYNCcnt), double-buffered, and shared by all 8 waves.
// Steady-state loop is branch-free (last K-step peeled); all 4 B operands are
// fetched from LDS before the 4 back-to-back WMMAs.
// mode 0: scatter qkv -> Q[B,H,N,64], K[B,H,N,64], VT[B,H,64,N]   (bf16)
// mode 1: outF = acc + bias[n] + res[m,n]                         (f32)
// mode 2: outB = bf16( gelu_exact(acc + bias[n]) )                (bf16)
// ---------------------------------------------------------------------------
#define BROW 40                 // padded LDS row stride (halves) for 32-half rows
__global__ void gemm_bf16_kernel(const bf16_t* __restrict__ A, const bf16_t* __restrict__ BT,
                                 const float* __restrict__ bias, const float* __restrict__ res,
                                 float* __restrict__ outF, bf16_t* __restrict__ outB,
                                 bf16_t* __restrict__ outQ, bf16_t* __restrict__ outK,
                                 bf16_t* __restrict__ outVT,
                                 int M, int Nout, int K, int mode) {
    __shared__ __align__(16) bf16_t Bs[2][64 * BROW];

    const int tid  = threadIdx.x;
    const int lane = tid & 31;
    const int wid  = tid >> 5;
    const int hi16 = lane >> 4;
    const int lm   = lane & 15;
    const int nT64 = Nout >> 6;
    const int bm   = blockIdx.x / nT64;
    const int bn   = blockIdx.x % nT64;
    const int m0   = bm * 128 + wid * 16;
    const int n0   = bn * 64;

    // per-thread slice of the 64x32 B tile copy: 256 threads x 16B = 4KB
    const int crow = tid >> 2;          // 0..63  (n within tile)
    const int cchk = tid & 3;           // 0..3   (8-half chunk within k-step)
    const bf16_t* gB = BT + (size_t)(n0 + crow) * K + cchk * 8;
    bf16_t* lB[2] = { &Bs[0][crow * BROW + cchk * 8], &Bs[1][crow * BROW + cchk * 8] };

    v8f c[4] = {};
    const bf16_t* arow = A + (size_t)(m0 + lm) * K;

    const int ksteps = K >> 5;
    copy16_g2l(gB, lB[0]);              // prologue prefetch (kk = 0)

    int p = 0;
    // ---- steady state: branch-free, always prefetching next tile ----
    for (int step = 0; step < ksteps - 1; ++step) {
        const int kk = step << 5;
        __syncthreads();                                    // done reading Bs[1-p]
        copy16_g2l(gB + (size_t)(kk + 32), lB[1 - p]);      // prefetch next tile
        wait_async_le1();                                   // Bs[p] DMA complete
        __syncthreads();                                    // all waves see Bs[p]

        __builtin_prefetch(arow + kk + 128, 0, 3);          // near-scope prefetch
        v16bf a = load_a16x32(arow, kk, hi16);
        v16bf bmat[4];
#pragma unroll
        for (int t = 0; t < 4; ++t) {
            const bf16_t* bp = &Bs[p][(t * 16 + lm) * BROW + 16 * hi16];
            bmat[t] = join16(*reinterpret_cast<const v8bf*>(bp),
                             *reinterpret_cast<const v8bf*>(bp + 8));
        }
#pragma unroll
        for (int t = 0; t < 4; ++t) c[t] = WMMA_BF16(a, bmat[t], c[t]);
        p ^= 1;
    }
    // ---- epilogue K-step (no further prefetch) ----
    {
        const int kk = (ksteps - 1) << 5;
        __syncthreads();
        wait_async_le0();
        __syncthreads();
        v16bf a = load_a16x32(arow, kk, hi16);
        v16bf bmat[4];
#pragma unroll
        for (int t = 0; t < 4; ++t) {
            const bf16_t* bp = &Bs[p][(t * 16 + lm) * BROW + 16 * hi16];
            bmat[t] = join16(*reinterpret_cast<const v8bf*>(bp),
                             *reinterpret_cast<const v8bf*>(bp + 8));
        }
#pragma unroll
        for (int t = 0; t < 4; ++t) c[t] = WMMA_BF16(a, bmat[t], c[t]);
    }

#pragma unroll
    for (int t = 0; t < 4; ++t) {
#pragma unroll
        for (int r = 0; r < 8; ++r) {
            const int m = m0 + r + 8 * hi16;     // C layout: row = r + 8*(L>>4)
            const int n = n0 + t * 16 + lm;      //           col = L&15
            float v = c[t][r];
            if (mode == 0) {
                const int bi   = m >> 11;        // m / N
                const int nq   = m & (Nn - 1);
                const int sect = n >> 10;        // 0:q 1:k 2:v
                const int cc   = n & 1023;
                const int h    = cc >> 6;
                const int d    = cc & 63;
                const bf16_t bv = f2bf(v);
                const size_t hb = (size_t)(bi * HEADS + h);
                if (sect == 0)      outQ[(hb * Nn + nq) * HD + d] = bv;
                else if (sect == 1) outK[(hb * Nn + nq) * HD + d] = bv;
                else                outVT[(hb * HD + d) * Nn + nq] = bv;
            } else if (mode == 1) {
                const size_t idx = (size_t)m * Nout + n;
                outF[idx] = v + bias[n] + res[idx];
            } else {
                float xg = v + bias[n];
                float gl = 0.5f * xg * (1.0f + erff(xg * 0.70710678118654752f));
                outB[(size_t)m * Nout + n] = f2bf(gl);
            }
        }
    }
}

// ---------------------------------------------------------------------------
// Flash attention: each wave owns 16 query rows of one (b,h).
//   Q [B,H,N,64] bf16, K [B,H,N,64] bf16, V stored transposed VT [B,H,64,N] bf16
//   O written as bf16 [B,N,DIM] (ready to be GEMM-A for the proj GEMM)
// Per 32-key step: 4 WMMAs for S, online softmax, P through LDS, 4 WMMAs for PV.
// ---------------------------------------------------------------------------
__global__ void attn_kernel(const bf16_t* __restrict__ Qb, const bf16_t* __restrict__ Kb,
                            const bf16_t* __restrict__ VTb, bf16_t* __restrict__ O) {
    __shared__ __align__(64) bf16_t pbuf[8][16 * 32];   // per-wave 16x32 P tile
    const int lane = threadIdx.x & 31;
    const int wid  = threadIdx.x >> 5;
    const int hi16 = lane >> 4;
    const int lm   = lane & 15;
    const int gw   = blockIdx.x * 8 + wid;
    const int bh   = gw >> 7;               // N/16 = 128 q-tiles per head
    const int q0   = (gw & 127) << 4;
    const int bIdx = bh >> 4;
    const int hIdx = bh & 15;

    const size_t headBase = (size_t)bh * Nn * HD;

    // Q as A operand: a0 covers d 0..31, a1 covers d 32..63
    const bf16_t* qrow = Qb + headBase + (size_t)(q0 + lm) * HD;
    const v16bf a0 = load_a16x32(qrow, 0, hi16);
    const v16bf a1 = load_a16x32(qrow, 32, hi16);

    v8f o0 = {}, o1 = {}, o2 = {}, o3 = {};
    float mrow[8], lrow[8];
#pragma unroll
    for (int r = 0; r < 8; ++r) { mrow[r] = -1e30f; lrow[r] = 0.f; }

    for (int kk = 0; kk < Nn; kk += 32) {
        // ---- S = Q K^T for 32 key columns (two 16-col tiles) ----
        v8f s0 = {}, s1v = {};
        {
            const bf16_t* kr0 = Kb + headBase + (size_t)(kk + lm) * HD + 16 * hi16;
            const bf16_t* kr1 = Kb + headBase + (size_t)(kk + 16 + lm) * HD + 16 * hi16;
            v16bf b00 = *reinterpret_cast<const v16bf*>(kr0);       // d 0..31 slice
            v16bf b01 = *reinterpret_cast<const v16bf*>(kr0 + 32);  // d 32..63 slice
            v16bf b10 = *reinterpret_cast<const v16bf*>(kr1);
            v16bf b11 = *reinterpret_cast<const v16bf*>(kr1 + 32);
            s0  = WMMA_BF16(a0, b00, s0);
            s0  = WMMA_BF16(a1, b01, s0);
            s1v = WMMA_BF16(a0, b10, s1v);
            s1v = WMMA_BF16(a1, b11, s1v);
        }

        // ---- online softmax over these 32 columns ----
#pragma unroll
        for (int r = 0; r < 8; ++r) {
            float v0 = s0[r] * 0.125f;         // scale = HEAD_DIM^-0.5
            float v1 = s1v[r] * 0.125f;
            float mx = fmaxf(v0, v1);
#pragma unroll
            for (int msk = 1; msk < 16; msk <<= 1) mx = fmaxf(mx, __shfl_xor(mx, msk, 32));
            float mn = fmaxf(mrow[r], mx);
            float al = __expf(mrow[r] - mn);
            float p0 = __expf(v0 - mn);
            float p1 = __expf(v1 - mn);
            float rs = p0 + p1;
#pragma unroll
            for (int msk = 1; msk < 16; msk <<= 1) rs += __shfl_xor(rs, msk, 32);
            lrow[r] = lrow[r] * al + rs;
            mrow[r] = mn;
            // C layout -> LDS row-major 16x32 P tile
            pbuf[wid][(r + 8 * hi16) * 32 + lm]      = f2bf(p0);
            pbuf[wid][(r + 8 * hi16) * 32 + 16 + lm] = f2bf(p1);
            o0[r] *= al; o1[r] *= al; o2[r] *= al; o3[r] *= al;
        }

        // ---- re-read P in A-operand layout (per-wave LDS, wave-synchronous) ----
        const bf16_t* pr = &pbuf[wid][lm * 32 + 16 * hi16];
        v16bf pa = join16(*reinterpret_cast<const v8bf*>(pr),
                          *reinterpret_cast<const v8bf*>(pr + 8));

        // ---- O += P @ V  (V transposed: contiguous over k) ----
        const bf16_t* vr = VTb + (size_t)bh * HD * Nn + (size_t)lm * Nn + kk + 16 * hi16;
        v16bf vb0 = *reinterpret_cast<const v16bf*>(vr);
        v16bf vb1 = *reinterpret_cast<const v16bf*>(vr + (size_t)16 * Nn);
        v16bf vb2 = *reinterpret_cast<const v16bf*>(vr + (size_t)32 * Nn);
        v16bf vb3 = *reinterpret_cast<const v16bf*>(vr + (size_t)48 * Nn);
        o0 = WMMA_BF16(pa, vb0, o0);
        o1 = WMMA_BF16(pa, vb1, o1);
        o2 = WMMA_BF16(pa, vb2, o2);
        o3 = WMMA_BF16(pa, vb3, o3);
    }

    // ---- normalize and scatter to [B,N,DIM] bf16 ----
#pragma unroll
    for (int r = 0; r < 8; ++r) {
        const float inv = 1.f / lrow[r];
        const int q = q0 + r + 8 * hi16;
        const size_t base = ((size_t)bIdx * Nn + q) * DIMd + hIdx * HD;
        O[base + 0  + lm] = f2bf(o0[r] * inv);
        O[base + 16 + lm] = f2bf(o1[r] * inv);
        O[base + 32 + lm] = f2bf(o2[r] * inv);
        O[base + 48 + lm] = f2bf(o3[r] * inv);
    }
}

// ---------------------------------------------------------------------------
// host-side orchestration
// ---------------------------------------------------------------------------
extern "C" void kernel_launch(void* const* d_in, const int* in_sizes, int n_in,
                              void* d_out, int out_size, void* d_ws, size_t ws_size,
                              hipStream_t stream) {
    (void)in_sizes; (void)n_in; (void)out_size; (void)ws_size;
    const float* x      = (const float*)d_in[0];
    const float* w_qkv  = (const float*)d_in[1];
    const float* w_proj = (const float*)d_in[2];
    const float* b_proj = (const float*)d_in[3];
    const float* ln1_g  = (const float*)d_in[4];
    const float* ln1_b  = (const float*)d_in[5];
    const float* w1     = (const float*)d_in[6];
    const float* b1     = (const float*)d_in[7];
    const float* w2     = (const float*)d_in[8];
    const float* b2     = (const float*)d_in[9];
    const float* ln2_g  = (const float*)d_in[10];
    const float* ln2_b  = (const float*)d_in[11];
    float* out = (float*)d_out;

    char* ws = (char*)d_ws;
    size_t cur = 0;
    auto alloc = [&](size_t bytes) -> char* {
        char* p = ws + cur;
        cur += (bytes + 255) & ~(size_t)255;
        return p;
    };
    bf16_t* h1     = (bf16_t*)alloc((size_t)Mrows * DIMd * 2);   // LN1 out
    bf16_t* wqkvT  = (bf16_t*)alloc((size_t)NQKV * DIMd * 2);
    bf16_t* wprojT = (bf16_t*)alloc((size_t)DIMd * DIMd * 2);
    bf16_t* w1T    = (bf16_t*)alloc((size_t)HID * DIMd * 2);
    bf16_t* w2T    = (bf16_t*)alloc((size_t)DIMd * HID * 2);
    bf16_t* Qb     = (bf16_t*)alloc((size_t)Mrows * DIMd * 2);
    bf16_t* Kb     = (bf16_t*)alloc((size_t)Mrows * DIMd * 2);
    bf16_t* VTb    = (bf16_t*)alloc((size_t)Mrows * DIMd * 2);
    bf16_t* obuf   = (bf16_t*)alloc((size_t)Mrows * DIMd * 2);
    float*  x1     = (float*) alloc((size_t)Mrows * DIMd * 4);   // post-attn residual
    bf16_t* h2     = (bf16_t*)alloc((size_t)Mrows * DIMd * 2);   // LN2 out
    bf16_t* h3     = (bf16_t*)alloc((size_t)Mrows * HID * 2);    // FFN1 out

    // weights -> bf16, transposed
    cvtT_kernel<<<(DIMd * NQKV) / 256, 256, 0, stream>>>(w_qkv, wqkvT, DIMd, NQKV);
    cvtT_kernel<<<(DIMd * DIMd) / 256, 256, 0, stream>>>(w_proj, wprojT, DIMd, DIMd);
    cvtT_kernel<<<(DIMd * HID) / 256, 256, 0, stream>>>(w1, w1T, DIMd, HID);
    cvtT_kernel<<<(HID * DIMd) / 256, 256, 0, stream>>>(w2, w2T, HID, DIMd);

    // LN1
    ln_bf16_kernel<<<Mrows, 256, 0, stream>>>(x, ln1_g, ln1_b, h1, DIMd);

    // QKV GEMM (mode 0: scatter q/k/vT as bf16)
    gemm_bf16_kernel<<<(Mrows / 128) * (NQKV / 64), 256, 0, stream>>>(
        h1, wqkvT, nullptr, nullptr, nullptr, nullptr, Qb, Kb, VTb,
        Mrows, NQKV, DIMd, 0);

    // attention (B*H*(N/16) wave-tiles, 8 waves per block)
    attn_kernel<<<(Bb * HEADS * (Nn / 16)) / 8, 256, 0, stream>>>(Qb, Kb, VTb, obuf);

    // proj GEMM + bias + residual(x)  -> x1 (f32)
    gemm_bf16_kernel<<<(Mrows / 128) * (DIMd / 64), 256, 0, stream>>>(
        obuf, wprojT, b_proj, x, x1, nullptr, nullptr, nullptr, nullptr,
        Mrows, DIMd, DIMd, 1);

    // LN2
    ln_bf16_kernel<<<Mrows, 256, 0, stream>>>(x1, ln2_g, ln2_b, h2, DIMd);

    // FFN1 + bias + exact GELU -> h3 (bf16)
    gemm_bf16_kernel<<<(Mrows / 128) * (HID / 64), 256, 0, stream>>>(
        h2, w1T, b1, nullptr, nullptr, h3, nullptr, nullptr, nullptr,
        Mrows, HID, DIMd, 2);

    // FFN2 + bias + residual(x1) -> out (f32)
    gemm_bf16_kernel<<<(Mrows / 128) * (DIMd / 64), 256, 0, stream>>>(
        h3, w2T, b2, x1, out, nullptr, nullptr, nullptr, nullptr,
        Mrows, DIMd, HID, 1);
}
